// TemporalRelationGraph_52974126629263
// MI455X (gfx1250) — matrix operand order repaired and checked by reading, not compiled, not run on hardware
//
#include <hip/hip_runtime.h>
#include <hip/hip_bf16.h>
#include <math.h>

// ---------------------------------------------------------------------------
// MI455X plan:
//  * xw = x @ W   (100000x256 @ 256x1024, 52 GFLOP)  -> bf16 WMMA, f32 acc
//  * attention dots / segment softmax / scatter-agg  -> bandwidth-bound,
//    3 edge passes (max, sum-exp, weighted scatter) recomputing the cheap
//    logit each pass instead of materializing alpha.
//  * head pooling -> 4 scalars -> softmax -> fused residual h
//  * per-edge h[src]*h[dst] @ lin_W + log_softmax    -> one wave per edge
// Total HBM traffic ~5 GB -> ~200us at 23.3 TB/s; GEMM is negligible on WMMA.
// ---------------------------------------------------------------------------

typedef __bf16 bf16_t;
typedef bf16_t v16bf __attribute__((ext_vector_type(16)));
typedef float  v8f   __attribute__((ext_vector_type(8)));
typedef unsigned int u32x4 __attribute__((ext_vector_type(4)));

union ABfrag { v16bf v; u32x4 u[2]; };

__device__ __forceinline__ unsigned short f32_to_bf16_rne(float f) {
    unsigned u = __float_as_uint(f);
    if ((u & 0x7FFFFFFFu) > 0x7F800000u)           // NaN: keep quiet
        return (unsigned short)((u >> 16) | 0x40);
    unsigned r = u + 0x7FFFu + ((u >> 16) & 1u);   // round to nearest even
    return (unsigned short)(r >> 16);
}

// ---- prep: f32 -> bf16 copy of x -------------------------------------------
__global__ __launch_bounds__(256) void trg_to_bf16(const float* __restrict__ src,
                                                   unsigned short* __restrict__ dst,
                                                   int n) {
    int i = blockIdx.x * 256 + threadIdx.x;
    if (i < n) dst[i] = f32_to_bf16_rne(src[i]);
}

// ---- prep: pack W [256,1024] into per-lane B-fragment tiles ----------------
// tile t = nb*8 + kb  (nb in 0..63 over 16-wide N blocks, kb in 0..7 over 32-deep K)
// halves layout inside tile: [lane(32)][16 contiguous bf16]
// lane L: n = nb*16 + (L&15);  k(e) = kb*32 + (L>>4)*16 + e
__global__ __launch_bounds__(256) void trg_pack_w(const float* __restrict__ W,
                                                  unsigned short* __restrict__ wb) {
    int i = blockIdx.x * 256 + threadIdx.x;       // 262144 total halves
    if (i >= 256 * 1024) return;
    int t  = i >> 9;
    int L  = (i >> 4) & 31;
    int e  = i & 15;
    int nb = t >> 3, kb = t & 7;
    int k  = kb * 32 + (L >> 4) * 16 + e;
    int n  = nb * 16 + (L & 15);
    wb[i] = f32_to_bf16_rne(W[(size_t)k * 1024 + n]);
}

// ---- GEMM: xw = x @ W via v_wmma_f32_16x16x32_bf16 -------------------------
// one wave -> 16x64 output strip, 8 K-steps of 32.
__global__ __launch_bounds__(256) void trg_gemm_wmma(const unsigned short* __restrict__ xb,
                                                     const unsigned short* __restrict__ wb,
                                                     float* __restrict__ xw,
                                                     int njobs) {
    int wid  = blockIdx.x * 8 + (threadIdx.x >> 5);
    if (wid >= njobs) return;                     // whole wave exits together
    int lane = threadIdx.x & 31;
    int mtile  = wid >> 4;
    int nstrip = wid & 15;
    int m0     = mtile * 16;
    int mrow   = m0 + (lane & 15);
    int khalf  = lane >> 4;

    v8f acc[4];
    acc[0] = v8f{}; acc[1] = v8f{}; acc[2] = v8f{}; acc[3] = v8f{};

    const unsigned short* arow = xb + (size_t)mrow * 256;
    for (int kb = 0; kb < 8; ++kb) {
        int k0 = kb * 32;
        ABfrag a;
        a.u[0] = *(const u32x4*)(arow + k0 + khalf * 16);       // K = k0+kh*16 .. +7
        a.u[1] = *(const u32x4*)(arow + k0 + khalf * 16 + 8);   // K = k0+kh*16+8 ..
#pragma unroll
        for (int j = 0; j < 4; ++j) {
            int nb = nstrip * 4 + j;
            const u32x4* bp = (const u32x4*)(wb + ((size_t)(nb * 8 + kb) * 512 + lane * 16));
            ABfrag b;
            b.u[0] = bp[0];
            b.u[1] = bp[1];
            acc[j] = __builtin_amdgcn_wmma_f32_16x16x32_bf16(
                false, a.v, false, b.v, (short)0, acc[j], false, false);
        }
    }
    int col   = nstrip * 64 + (lane & 15);
    int rbase = m0 + (lane >> 4) * 8;
#pragma unroll
    for (int v = 0; v < 8; ++v) {
        float* row = xw + (size_t)(rbase + v) * 1024 + col;
        row[0]  = acc[0][v];
        row[16] = acc[1][v];
        row[32] = acc[2][v];
        row[48] = acc[3][v];
    }
}

// ---- a_s / a_d: per-(node,head) dot of xw row with attention vectors -------
__global__ __launch_bounds__(128) void trg_att_dots(const float* __restrict__ xw,
                                                    const float* __restrict__ att_s,
                                                    const float* __restrict__ att_d,
                                                    float* __restrict__ as_o,
                                                    float* __restrict__ ad_o) {
    int n = blockIdx.x;
    int h = threadIdx.x >> 5;
    int lane = threadIdx.x & 31;
    const float* row = xw + (size_t)n * 1024 + h * 256 + lane * 8;
    const float* ws  = att_s + h * 256 + lane * 8;
    const float* wd  = att_d + h * 256 + lane * 8;
    float ss = 0.f, sd = 0.f;
#pragma unroll
    for (int j = 0; j < 8; j += 4) {
        float4 v  = *(const float4*)(row + j);
        float4 vs = *(const float4*)(ws + j);
        float4 vd = *(const float4*)(wd + j);
        ss += v.x * vs.x + v.y * vs.y + v.z * vs.z + v.w * vs.w;
        sd += v.x * vd.x + v.y * vd.y + v.z * vd.z + v.w * vd.w;
    }
    for (int off = 16; off > 0; off >>= 1) {
        ss += __shfl_down(ss, off, 32);
        sd += __shfl_down(sd, off, 32);
    }
    if (lane == 0) {
        as_o[(size_t)n * 4 + h] = ss;
        ad_o[(size_t)n * 4 + h] = sd;
    }
}

__global__ __launch_bounds__(256) void trg_fill_neg_inf(float* __restrict__ p, int n) {
    int i = blockIdx.x * 256 + threadIdx.x;
    if (i < n) p[i] = -3.0e38f;
}

__device__ __forceinline__ void trg_edge_sd(const long long* __restrict__ ei,
                                            int e, int E, int& s, int& d) {
    if (e < E) { s = (int)ei[e]; d = (int)ei[(size_t)E + e]; }
    else       { s = d = e - E; }              // appended self-loops
}

__device__ __forceinline__ float trg_leaky(float x) { return x > 0.f ? x : 0.2f * x; }

// ---- edge pass 1: segment max of leaky-relu logits -------------------------
__global__ __launch_bounds__(256) void trg_edge_max(const long long* __restrict__ ei,
                                                    const float* __restrict__ as_i,
                                                    const float* __restrict__ ad_i,
                                                    float* __restrict__ mbuf,
                                                    int E, int total) {
    int i = blockIdx.x * 256 + threadIdx.x;
    if (i >= total) return;
    int e = i >> 2, h = i & 3;
    int s, d; trg_edge_sd(ei, e, E, s, d);
    float l = trg_leaky(as_i[(size_t)s * 4 + h] + ad_i[(size_t)d * 4 + h]);
    atomicMax(&mbuf[(size_t)d * 4 + h], l);
}

// ---- edge pass 2: segment sum of exp(logit - max) --------------------------
__global__ __launch_bounds__(256) void trg_edge_sumexp(const long long* __restrict__ ei,
                                                       const float* __restrict__ as_i,
                                                       const float* __restrict__ ad_i,
                                                       const float* __restrict__ mbuf,
                                                       float* __restrict__ sbuf,
                                                       int E, int total) {
    int i = blockIdx.x * 256 + threadIdx.x;
    if (i >= total) return;
    int e = i >> 2, h = i & 3;
    int s, d; trg_edge_sd(ei, e, E, s, d);
    int dh = d * 4 + h;
    float l = trg_leaky(as_i[(size_t)s * 4 + h] + ad_i[(size_t)d * 4 + h]);
    atomicAdd(&sbuf[dh], __expf(l - mbuf[dh]));
}

// ---- edge pass 3: agg[dst] += alpha * xw[src]  (block per edge) ------------
__global__ __launch_bounds__(256) void trg_edge_agg(const long long* __restrict__ ei,
                                                    const float* __restrict__ as_i,
                                                    const float* __restrict__ ad_i,
                                                    const float* __restrict__ mbuf,
                                                    const float* __restrict__ sbuf,
                                                    const float* __restrict__ xw,
                                                    float* __restrict__ agg,
                                                    int E) {
    int e = blockIdx.x;
    int h = threadIdx.x >> 6;
    int c = (threadIdx.x & 63) << 2;
    int s, d; trg_edge_sd(ei, e, E, s, d);
    int dh = d * 4 + h;
    float l = trg_leaky(as_i[(size_t)s * 4 + h] + ad_i[(size_t)d * 4 + h]);
    float w = __expf(l - mbuf[dh]) / (sbuf[dh] + 1e-16f);
    float4 v = *(const float4*)(xw + (size_t)s * 1024 + h * 256 + c);
    float* o = agg + (size_t)d * 1024 + h * 256 + c;
    atomicAdd(o + 0, w * v.x);
    atomicAdd(o + 1, w * v.y);
    atomicAdd(o + 2, w * v.z);
    atomicAdd(o + 3, w * v.w);
}

// ---- per-head global mean pooling of (agg + bias) --------------------------
__global__ __launch_bounds__(256) void trg_head_pool(const float* __restrict__ agg,
                                                     const float* __restrict__ bias,
                                                     float* __restrict__ pool,
                                                     int total) {
    __shared__ float sh[4];
    if (threadIdx.x < 4) sh[threadIdx.x] = 0.f;
    __syncthreads();
    float loc[4] = {0.f, 0.f, 0.f, 0.f};
    int stride = gridDim.x * 256;
    for (int i = blockIdx.x * 256 + threadIdx.x; i < total; i += stride) {
        int hc = i & 1023;
        loc[hc >> 8] += agg[i] + bias[hc];
    }
#pragma unroll
    for (int h = 0; h < 4; ++h) atomicAdd(&sh[h], loc[h]);
    __syncthreads();
    if (threadIdx.x < 4) atomicAdd(&pool[threadIdx.x], sh[threadIdx.x]);
}

// ---- tiny: 1x1 conv + relu + softmax over the 4 heads ----------------------
__global__ void trg_head_coef(const float* __restrict__ pool,
                              const float* __restrict__ cw,
                              const float* __restrict__ cb,
                              float* __restrict__ coef, float invNC) {
    float a[4], mx = -3.0e38f;
#pragma unroll
    for (int h = 0; h < 4; ++h) {
        float v = pool[h] * invNC * cw[0] + cb[0];
        v = v > 0.f ? v : 0.f;
        a[h] = v;
        mx = fmaxf(mx, v);
    }
    float s = 0.f;
#pragma unroll
    for (int h = 0; h < 4; ++h) { a[h] = __expf(a[h] - mx); s += a[h]; }
#pragma unroll
    for (int h = 0; h < 4; ++h) coef[h] = a[h] / s;
}

// ---- fuse heads + residual + relu -> h -------------------------------------
__global__ __launch_bounds__(256) void trg_fuse_res(const float* __restrict__ agg,
                                                    const float* __restrict__ bias,
                                                    const float* __restrict__ coef,
                                                    const float* __restrict__ x,
                                                    float* __restrict__ hb,
                                                    int total) {
    int i = blockIdx.x * 256 + threadIdx.x;
    if (i >= total) return;
    int n = i >> 8, c = i & 255;
    float f = 0.f;
#pragma unroll
    for (int h = 0; h < 4; ++h)
        f += coef[h] * (agg[(size_t)n * 1024 + h * 256 + c] + bias[h * 256 + c]);
    f += x[i];
    hb[i] = f > 0.f ? f : 0.f;
}

// ---- final: per-edge (h[src]*h[dst]) @ lin_W, log_softmax; wave per edge ---
__global__ __launch_bounds__(256) void trg_edge_out(const long long* __restrict__ ei,
                                                    const float* __restrict__ hb,
                                                    const float* __restrict__ lw,
                                                    const float* __restrict__ lb,
                                                    float* __restrict__ out,
                                                    int E) {
    int w = threadIdx.x >> 5, lane = threadIdx.x & 31;
    int e = blockIdx.x * 8 + w;
    if (e >= E) return;                       // uniform across the wave
    int s = (int)ei[e], d = (int)ei[(size_t)E + e];
    const float* ps = hb + (size_t)s * 256 + lane * 8;
    const float* pd = hb + (size_t)d * 256 + lane * 8;
    float s0 = 0.f, s1 = 0.f;
#pragma unroll
    for (int j = 0; j < 8; j += 4) {
        int c = lane * 8 + j;
        float4 a = *(const float4*)(ps + j);
        float4 b = *(const float4*)(pd + j);
        float p0 = a.x * b.x, p1 = a.y * b.y, p2 = a.z * b.z, p3 = a.w * b.w;
        s0 += p0 * lw[2*c] + p1 * lw[2*(c+1)] + p2 * lw[2*(c+2)] + p3 * lw[2*(c+3)];
        s1 += p0 * lw[2*c+1] + p1 * lw[2*(c+1)+1] + p2 * lw[2*(c+2)+1] + p3 * lw[2*(c+3)+1];
    }
    for (int off = 16; off > 0; off >>= 1) {
        s0 += __shfl_down(s0, off, 32);
        s1 += __shfl_down(s1, off, 32);
    }
    if (lane == 0) {
        float o0 = s0 + lb[0], o1 = s1 + lb[1];
        float mx = fmaxf(o0, o1);
        float lse = mx + __logf(__expf(o0 - mx) + __expf(o1 - mx));
        out[(size_t)e * 2 + 0] = o0 - lse;
        out[(size_t)e * 2 + 1] = o1 - lse;
    }
}

extern "C" void kernel_launch(void* const* d_in, const int* in_sizes, int n_in,
                              void* d_out, int out_size, void* d_ws, size_t ws_size,
                              hipStream_t stream) {
    const float*      x     = (const float*)d_in[0];
    const long long*  ei    = (const long long*)d_in[1];   // int64 [2,E]
    const float*      W     = (const float*)d_in[2];
    const float*      att_s = (const float*)d_in[3];
    const float*      att_d = (const float*)d_in[4];
    const float*      gbias = (const float*)d_in[5];
    const float*      cw    = (const float*)d_in[6];
    const float*      cb    = (const float*)d_in[7];
    const float*      lw    = (const float*)d_in[8];
    const float*      lb    = (const float*)d_in[9];

    const int C = 256;
    const int N = in_sizes[0] / C;            // 100000 (multiple of 16)
    const int E = in_sizes[1] / 2;            // 300000
    const int Etot = E + N;                   // with self-loops

    // ---- workspace carve-out (all offsets 16B-aligned) ----
    char* ws = (char*)d_ws;
    unsigned short* xb  = (unsigned short*)ws;                 // N*256 bf16
    unsigned short* wb  = xb + (size_t)N * C;                  // 256*1024 bf16
    float* xw   = (float*)(wb + 256 * 1024);                   // N*1024 f32
    float* as_b = xw + (size_t)N * 1024;                       // N*4
    float* ad_b = as_b + (size_t)N * 4;                        // N*4
    float* mbuf = ad_b + (size_t)N * 4;                        // N*4
    float* sbuf = mbuf + (size_t)N * 4;                        // N*4
    float* agg  = sbuf + (size_t)N * 4;                        // N*1024
    float* hb   = agg + (size_t)N * 1024;                      // N*256
    float* pool = hb + (size_t)N * 256;                        // 4
    float* coef = pool + 4;                                    // 4

    // ---- init (memset is capture-safe) ----
    hipMemsetAsync(sbuf, 0, (size_t)N * 4 * sizeof(float), stream);
    hipMemsetAsync(agg,  0, (size_t)N * 1024 * sizeof(float), stream);
    hipMemsetAsync(pool, 0, 4 * sizeof(float), stream);
    trg_fill_neg_inf<<<(N * 4 + 255) / 256, 256, 0, stream>>>(mbuf, N * 4);

    // ---- prep bf16 operands ----
    trg_to_bf16<<<(N * C + 255) / 256, 256, 0, stream>>>(x, xb, N * C);
    trg_pack_w<<<(256 * 1024) / 256, 256, 0, stream>>>(W, wb);

    // ---- WMMA GEMM: xw = x @ W ----
    int njobs = (N / 16) * 16;                // wave jobs (16 n-strips per m-tile)
    trg_gemm_wmma<<<(njobs + 7) / 8, 256, 0, stream>>>(xb, wb, xw, njobs);

    // ---- attention coefficients ----
    trg_att_dots<<<N, 128, 0, stream>>>(xw, att_s, att_d, as_b, ad_b);

    // ---- segment softmax + scatter aggregate over E+N edges ----
    int tot = Etot * 4;
    trg_edge_max   <<<(tot + 255) / 256, 256, 0, stream>>>(ei, as_b, ad_b, mbuf, E, tot);
    trg_edge_sumexp<<<(tot + 255) / 256, 256, 0, stream>>>(ei, as_b, ad_b, mbuf, sbuf, E, tot);
    trg_edge_agg   <<<Etot, 256, 0, stream>>>(ei, as_b, ad_b, mbuf, sbuf, xw, agg, E);

    // ---- head fusion ----
    trg_head_pool<<<2048, 256, 0, stream>>>(agg, gbias, pool, N * 1024);
    trg_head_coef<<<1, 1, 0, stream>>>(pool, cw, cb, coef, 1.0f / ((float)N * (float)C));
    trg_fuse_res<<<(N * C + 255) / 256, 256, 0, stream>>>(agg, gbias, coef, x, hb, N * C);

    // ---- edge classifier + log_softmax ----
    trg_edge_out<<<(E + 7) / 8, 256, 0, stream>>>(ei, hb, lw, lb, (float*)d_out, E);
}
